// QuadraticNetCholesky_76347338654404
// MI455X (gfx1250) — matrix operand order repaired and checked by reading, not compiled, not run on hardware
//
#include <hip/hip_runtime.h>

// CDNA5 / gfx1250, wave32.
// Prep kernel packs f16 weights once into d_ws; main kernel DMAs them to LDS
// via the Tensor Data Mover, runs the MLP on V_WMMA_F32_16X16X32_F16, and
// finishes with y = ||L^T x||^2 (never materializes A = L L^T).

typedef __attribute__((ext_vector_type(16))) _Float16     v16h;
typedef __attribute__((ext_vector_type(8)))  float        v8f;
typedef __attribute__((ext_vector_type(4)))  unsigned int u32x4;
typedef __attribute__((ext_vector_type(8)))  int          i32x8;
typedef __attribute__((ext_vector_type(4)))  int          i32x4;

union FragH {
  v16h  h;
  uint4 q[2];
};

// ---- packed weight blob layout in d_ws (all 16B aligned) ----
#define OFF_W1   0u       // f16 [128][32]  (K 24->32 pad)      8192 B
#define OFF_W2   8192u    // f16 [32][128]                      8192 B
#define OFF_W3   16384u   // f16 [304][32]  (N 300->304 pad)   19456 B
#define OFF_B1   35840u   // f32 [128]                           512 B
#define OFF_B2   36352u   // f32 [32]                            128 B
#define OFF_B3   36480u   // f32 [304]                          1216 B
#define BLOB_BYTES 37696u
#define BLOB_DWORDS 9424u

static __device__ __forceinline__ v8f wmma_f32_f16(v16h a, v16h b, v8f c) {
  return __builtin_amdgcn_wmma_f32_16x16x32_f16(false, a, false, b, (short)0, c, false, false);
}
static __device__ __forceinline__ float elu_f(float t) {
  return t > 0.0f ? t : (__expf(t) - 1.0f);
}
static __device__ __forceinline__ float softplus_f(float t) {
  // v_exp_f32 + v_log_f32 fast path; 1+e^t >= 1 so no cancellation (log1p not needed)
  return t > 20.0f ? t : __logf(1.0f + __expf(t));
}

// ======================= prep: pack weights once =======================
__global__ __launch_bounds__(256)
void qnet_prep_kernel(const float* __restrict__ W1, const float* __restrict__ b1,
                      const float* __restrict__ W2, const float* __restrict__ b2,
                      const float* __restrict__ W3, const float* __restrict__ b3,
                      unsigned char* __restrict__ ws)
{
  _Float16* W1h = (_Float16*)(ws + OFF_W1);
  _Float16* W2h = (_Float16*)(ws + OFF_W2);
  _Float16* W3h = (_Float16*)(ws + OFF_W3);
  float* B1 = (float*)(ws + OFF_B1);
  float* B2 = (float*)(ws + OFF_B2);
  float* B3 = (float*)(ws + OFF_B3);
  int t = blockIdx.x * blockDim.x + threadIdx.x;
  if (t < 4096) {                               // W1: [128][24] -> [128][32] f16
    int n = t >> 5, k = t & 31;
    W1h[t] = (_Float16)((k < 24) ? W1[n * 24 + k] : 0.0f);
  } else if (t < 8192) {                        // W2: [32][128] f16
    int i = t - 4096;
    W2h[i] = (_Float16)W2[i];
  } else if (t < 17920) {                       // W3: [300][32] -> [304][32] f16
    int i = t - 8192;
    W3h[i] = (_Float16)(((i >> 5) < 300) ? W3[i] : 0.0f);
  } else if (t < 18048) {
    int i = t - 17920; B1[i] = b1[i];
  } else if (t < 18080) {
    int i = t - 18048; B2[i] = b2[i];
  } else if (t < 18384) {
    int i = t - 18080; B3[i] = (i < 300) ? b3[i] : 0.0f;
  }
}

// ======================= main kernel =======================
#define TILES 4   // 16-row tiles per workgroup -> 64 rows/block

__global__ __launch_bounds__(32)
void qnet_main_kernel(const float* __restrict__ x,
                      const unsigned char* __restrict__ wblob,
                      float* __restrict__ out, int nRows)
{
  __shared__ __align__(16) unsigned char sBlob[BLOB_BYTES];
  __shared__ __align__(16) _Float16 sXh[16 * 32];
  __shared__ __align__(16) float    sXf[16 * 24];
  __shared__ __align__(16) _Float16 sH1[16 * 128];
  __shared__ __align__(16) _Float16 sH2[16 * 32];
  __shared__ __align__(16) float    sC [16 * 304];
  __shared__ __align__(16) float    sY [32];

  const _Float16* sW1 = (const _Float16*)(sBlob + OFF_W1);
  const _Float16* sW2 = (const _Float16*)(sBlob + OFF_W2);
  const _Float16* sW3 = (const _Float16*)(sBlob + OFF_W3);
  const float* sB1 = (const float*)(sBlob + OFF_B1);
  const float* sB2 = (const float*)(sBlob + OFF_B2);
  const float* sB3 = (const float*)(sBlob + OFF_B3);

  const int lane = threadIdx.x;               // one wave per workgroup

  // ---- weight blob: L2 -> LDS, bulk, no VALU ----
#if __has_builtin(__builtin_amdgcn_tensor_load_to_lds) && defined(__clang_major__) && (__clang_major__ <= 22)
  {
    // Tensor DMA descriptor (D#), CDNA5 ISA:
    // 1-row 2D tile of 9424 dwords; data_size=4B; no pad/iterate/multicast.
    unsigned long long ga = (unsigned long long)(uintptr_t)wblob;
    unsigned int ldsa = (unsigned int)(uintptr_t)(void*)sBlob;
    u32x4 g0;
    g0[0] = 1u;                                          // count=1, user desc
    g0[1] = ldsa;                                        // lds_addr
    g0[2] = (unsigned int)(ga & 0xFFFFFFFFu);            // global_addr[31:0]
    g0[3] = (unsigned int)(((ga >> 32) & 0x01FFFFFFu) | 0x80000000u); // ga[56:32] | type=2
    i32x8 g1;
    g1[0] = 0x00020000;                                  // data_size = 4 bytes
    g1[1] = (int)(BLOB_DWORDS << 16);                    // tensor_dim0[15:0]
    g1[2] = (int)(1u << 16);                             // tensor_dim1 = 1
    g1[3] = (int)(BLOB_DWORDS << 16);                    // tile_dim0
    g1[4] = 1;                                           // tile_dim1 = 1
    g1[5] = (int)BLOB_DWORDS;                            // tensor_dim0_stride
    g1[6] = 0;
    g1[7] = 0;
    i32x4 gz; gz[0] = 0; gz[1] = 0; gz[2] = 0; gz[3] = 0;
    __builtin_amdgcn_tensor_load_to_lds(g0, g1, gz, gz, 0);
    __builtin_amdgcn_s_wait_tensorcnt(0);
  }
#else
  {
    const uint4* src = (const uint4*)wblob;
    uint4* dst = (uint4*)sBlob;
    for (int i = lane; i < (int)(BLOB_BYTES / 16); i += 32) dst[i] = src[i];
  }
#endif
  __syncthreads();

  const int mrow = lane & 15;   // A row M / B col N / D col N
  const int hi   = lane >> 4;   // 0 = low-K half lanes, 1 = high-K half

#pragma clang loop unroll(disable)
  for (int t = 0; t < TILES; ++t) {
    const int rowBase = blockIdx.x * (16 * TILES) + t * 16;

    // ---- stage x tile: f32 copy + padded f16 (clamped rows keep EXEC uniform) ----
    for (int i = lane; i < 16 * 32; i += 32) {
      int r = i >> 5, k = i & 31;
      float v = 0.0f;
      if (k < 24) {
        int gr = rowBase + r; if (gr >= nRows) gr = nRows - 1;
        v = x[gr * 24 + k];
        sXf[r * 24 + k] = v;
      }
      sXh[i] = (_Float16)v;
    }
    __syncthreads();

    // ============ GEMM1: [16x32] x [32x128] -> h1 [16x128] ============
    FragH ax;
    {
      const _Float16* p = &sXh[(mrow << 5) + (hi << 3)];
      ax.q[0] = *reinterpret_cast<const uint4*>(p);       // K = kb .. kb+7
      ax.q[1] = *reinterpret_cast<const uint4*>(p + 16);  // K = kb+16 .. kb+23
    }
    for (int nt = 0; nt < 8; ++nt) {
      FragH bw;
      const _Float16* pb = &sW1[(((nt << 4) + mrow) << 5) + (hi << 3)];
      bw.q[0] = *reinterpret_cast<const uint4*>(pb);
      bw.q[1] = *reinterpret_cast<const uint4*>(pb + 16);
      v8f acc = {};
      acc = wmma_f32_f16(ax.h, bw.h, acc);
      const int n = (nt << 4) + mrow;
      const float bias = sB1[n];
#pragma unroll
      for (int v = 0; v < 8; ++v) {
        int m = v + (hi << 3);                            // D: M = vgpr + 8*hi
        sH1[(m << 7) + n] = (_Float16)elu_f(acc[v] + bias);
      }
    }
    __syncthreads();

    // ============ GEMM2: [16x128] x [128x32] -> h2 [16x32] ============
    v8f acc2[2] = {v8f{}, v8f{}};
    for (int kc = 0; kc < 4; ++kc) {
      FragH a2;
      const _Float16* p = &sH1[(mrow << 7) + (kc << 5) + (hi << 3)];
      a2.q[0] = *reinterpret_cast<const uint4*>(p);
      a2.q[1] = *reinterpret_cast<const uint4*>(p + 16);
#pragma unroll
      for (int nt = 0; nt < 2; ++nt) {
        FragH bw;
        const _Float16* pb = &sW2[(((nt << 4) + mrow) << 7) + (kc << 5) + (hi << 3)];
        bw.q[0] = *reinterpret_cast<const uint4*>(pb);
        bw.q[1] = *reinterpret_cast<const uint4*>(pb + 16);
        acc2[nt] = wmma_f32_f16(a2.h, bw.h, acc2[nt]);
      }
    }
#pragma unroll
    for (int nt = 0; nt < 2; ++nt) {
      const int n = (nt << 4) + mrow;
      const float bias = sB2[n];
#pragma unroll
      for (int v = 0; v < 8; ++v) {
        int m = v + (hi << 3);
        sH2[(m << 5) + n] = (_Float16)elu_f(acc2[nt][v] + bias);
      }
    }
    __syncthreads();

    // ======= GEMM3: [16x32] x [32x304] -> softplus c [16x304] =======
    FragH a3;
    {
      const _Float16* p = &sH2[(mrow << 5) + (hi << 3)];
      a3.q[0] = *reinterpret_cast<const uint4*>(p);
      a3.q[1] = *reinterpret_cast<const uint4*>(p + 16);
    }
    for (int nt = 0; nt < 19; ++nt) {
      FragH bw;
      const _Float16* pb = &sW3[(((nt << 4) + mrow) << 5) + (hi << 3)];
      bw.q[0] = *reinterpret_cast<const uint4*>(pb);
      bw.q[1] = *reinterpret_cast<const uint4*>(pb + 16);
      v8f d = {};
      d = wmma_f32_f16(a3.h, bw.h, d);
      const int n = (nt << 4) + mrow;
      const float bias = sB3[n];
#pragma unroll
      for (int v = 0; v < 8; ++v) {
        int m = v + (hi << 3);
        sC[m * 304 + n] = softplus_f(d[v] + bias);
      }
    }
    __syncthreads();

    // ==== y = ||L^T x||^2 = sum_j (sum_{i>=j} x_i L[i,j])^2 ; 2 lanes/row ====
    {
      const int r = lane >> 1, half = lane & 1;
      const float* xr = &sXf[r * 24];
      const float* cr = &sC[r * 304];
      float accq = 0.0f;
      for (int j = half; j < 24; j += 2) {
        float vsum = 0.0f;
        for (int i = j; i < 24; ++i)
          vsum += xr[i] * cr[((i * (i + 1)) >> 1) + j];   // tril idx(i,j)
        accq += vsum * vsum;
      }
      sY[lane] = accq;
    }
    __syncthreads();
    if (lane < 16) {
      int gr = rowBase + lane;
      if (gr < nRows) out[gr] = sY[2 * lane] + sY[2 * lane + 1];
    }
    __syncthreads();   // LDS reused by next tile
  }
}

extern "C" void kernel_launch(void* const* d_in, const int* in_sizes, int n_in,
                              void* d_out, int out_size, void* d_ws, size_t ws_size,
                              hipStream_t stream) {
  (void)n_in; (void)out_size; (void)ws_size;
  const float* x  = (const float*)d_in[0];
  const float* W1 = (const float*)d_in[1];
  const float* b1 = (const float*)d_in[2];
  const float* W2 = (const float*)d_in[3];
  const float* b2 = (const float*)d_in[4];
  const float* W3 = (const float*)d_in[5];
  const float* b3 = (const float*)d_in[6];
  float* out = (float*)d_out;
  unsigned char* ws = (unsigned char*)d_ws;

  hipLaunchKernelGGL(qnet_prep_kernel, dim3(72), dim3(256), 0, stream,
                     W1, b1, W2, b2, W3, b3, ws);

  const int nRows = in_sizes[0] / 24;
  const int grid  = (nRows + (16 * TILES) - 1) / (16 * TILES);
  hipLaunchKernelGGL(qnet_main_kernel, dim3(grid), dim3(32), 0, stream,
                     x, ws, out, nRows);
}
